// LSTM_5385888989323
// MI455X (gfx1250) — matrix operand order repaired and checked by reading, not compiled, not run on hardware
//
#include <hip/hip_runtime.h>

// LSTM: SEQ=512, BATCH=64, IN=256, HID=1024, 4 gates (f,i,o,cand)
// gate g uses W_{2g+1} (x-side), W_{2g+2} (h-side), biases b_{2g+1}+b_{2g+2}.

#define SEQ   512
#define BATCH 64
#define ISZ   256
#define HID   1024

typedef unsigned short u16;
typedef __attribute__((ext_vector_type(16))) __bf16 v16bf;
typedef __attribute__((ext_vector_type(8)))  float  v8f;
typedef __attribute__((ext_vector_type(4)))  unsigned int v4u;
typedef __attribute__((ext_vector_type(8)))  int v8i;
typedef __attribute__((ext_vector_type(4)))  int v4i;

#if __has_builtin(__builtin_amdgcn_tensor_load_to_lds) && \
    __has_builtin(__builtin_amdgcn_s_wait_tensorcnt)
#define HAVE_TDM 1
#else
#define HAVE_TDM 0
#endif

// ---------------- helpers ----------------

__device__ __forceinline__ u16 f2bf(float f) {
    unsigned int u = __float_as_uint(f);
    unsigned int r = (u + 0x7FFFu + ((u >> 16) & 1u)) >> 16;  // RNE
    return (u16)r;
}

__device__ __forceinline__ float fsig(float x) {
    return 1.0f / (1.0f + __expf(-x));
}

// Load a 32-byte WMMA fragment from two 16-byte-aligned chunks (global or LDS).
__device__ __forceinline__ v16bf ld_frag2(const u16* p0, const u16* p1) {
    union { uint4 u[2]; v16bf v; } t;
    t.u[0] = *(const uint4*)p0;
    t.u[1] = *(const uint4*)p1;
    return t.v;
}

#if HAVE_TDM
// One-shot 1-D TDM copy: global -> LDS, `bytes` must be a multiple of 8,
// bytes/8 <= 65535. Issued by one wave; tracked with TENSORcnt.
__device__ __forceinline__ void tdm_load_1d(const void* gsrc, void* ldst,
                                            unsigned bytes) {
    unsigned long long ga = (unsigned long long)gsrc;
    unsigned la = (unsigned)(unsigned long long)ldst;   // LDS aperture: addr[31:0]
    unsigned n8 = bytes >> 3;                           // 8-byte elements
    v4u g0;
    g0.x = 1u;                                          // count=1 (valid D#)
    g0.y = la;                                          // lds_addr
    g0.z = (unsigned)(ga & 0xFFFFFFFFu);                // global_addr[31:0]
    g0.w = (unsigned)((ga >> 32) & 0x1FFFFFFu) | (2u << 30);  // ga[56:32]|type=2
    v8i g1;
    g1[0] = (int)(3u << 16);                 // workgroup_mask=0, data_size=3 (8B)
    g1[1] = (int)((n8 & 0xFFFFu) << 16);     // tensor_dim0[15:0]
    g1[2] = (int)(((n8 >> 16) & 0xFFFFu) | (1u << 16)); // tdim0[31:16], tdim1=1
    g1[3] = (int)((n8 & 0xFFFFu) << 16);     // tile_dim0 (tdim1[31:16]=0)
    g1[4] = 0;                               // tile_dim1=0 (unused), tile_dim2=0
    g1[5] = (int)n8;                         // tensor_dim0_stride[31:0]
    g1[6] = 0;
    g1[7] = 0;
    v4i z4 = {0, 0, 0, 0};
    v8i z8 = {0, 0, 0, 0, 0, 0, 0, 0};
    // 6-arg toolchain form: (g0, g1, g2, g3, extra, cpol)
    __builtin_amdgcn_tensor_load_to_lds(g0, g1, z4, z4, z8, 0);
}
#endif

// ---------------- one-time prep kernels ----------------

// Pack W (K x 1024 f32, row-major) per gate into fragment-friendly bf16:
// dst[n*K + k] = W_g[k*HID + c], n = g*1024 + c  (N-major, K contiguous).
__global__ void k_pack(const float* __restrict__ A0, const float* __restrict__ A1,
                       const float* __restrict__ A2, const float* __restrict__ A3,
                       u16* __restrict__ dst, int K) {
    int idx = blockIdx.x * blockDim.x + threadIdx.x;   // exactly 4096*K threads
    int n = idx / K;
    int k = idx - n * K;
    int g = n >> 10;
    int c = n & 1023;
    const float* A = (g == 0) ? A0 : (g == 1) ? A1 : (g == 2) ? A2 : A3;
    dst[idx] = f2bf(A[(size_t)k * HID + c]);
}

__global__ void k_cvt_x(const float* __restrict__ src, u16* __restrict__ dst) {
    int i = blockIdx.x * blockDim.x + threadIdx.x;     // SEQ*BATCH*ISZ threads
    dst[i] = f2bf(src[i]);
}

// bias[n] = bx_g[c] + bh_g[c]; zero cell state and h(bf16) ping buffer.
__global__ void k_prep(const float* __restrict__ bx0, const float* __restrict__ bh0,
                       const float* __restrict__ bx1, const float* __restrict__ bh1,
                       const float* __restrict__ bx2, const float* __restrict__ bh2,
                       const float* __restrict__ bx3, const float* __restrict__ bh3,
                       float* __restrict__ bias, float* __restrict__ Cst,
                       u16* __restrict__ hb0) {
    int i = blockIdx.x * blockDim.x + threadIdx.x;     // BATCH*HID = 65536 threads
    if (i < 4 * HID) {
        int g = i >> 10, c = i & 1023;
        const float* bx = (g == 0) ? bx0 : (g == 1) ? bx1 : (g == 2) ? bx2 : bx3;
        const float* bh = (g == 0) ? bh0 : (g == 1) ? bh1 : (g == 2) ? bh2 : bh3;
        bias[i] = bx[c] + bh[c];
    }
    Cst[i] = 0.0f;
    hb0[i] = 0;
}

// ---------------- per-timestep kernel ----------------
//
// grid.x = 64 (HID/16 column strips), block = 512 threads = 16 waves.
// wave = (mtile 0..3, gate 0..3); each wave owns one 16x16 C tile:
//   rows = batch 16*mtile..+15, cols = gate*1024 + c0..+15 (c0 = 16*blockIdx.x).
// Stage: TDM copies h_{t-1} (128KB) and x_t (32KB) into LDS; inner loop reads
// A-fragments from LDS (ds_load_b128) and packed-B weights from L2 (b128).
// K = 256 (x) + 1024 (h) = 40 k-tiles of 32, bf16 WMMA, f32 accumulate.

#define LDS_DYN_BYTES ((BATCH * HID + BATCH * ISZ) * 2)   // 160 KB

__global__ __launch_bounds__(512) void lstm_step(
    const u16*  __restrict__ xb,    // (64,256)  bf16, this timestep
    const u16*  __restrict__ WxP,   // (4096,256) packed bf16
    const u16*  __restrict__ WhP,   // (4096,1024) packed bf16
    const float* __restrict__ bias, // (4096)
    const u16*  __restrict__ hin,   // (64,1024) bf16
    u16*        __restrict__ hout,  // (64,1024) bf16
    float*      __restrict__ Cst,   // (64,1024)
    float*      __restrict__ hseq,  // (64,1024) f32 -> d_out slice
    float*      __restrict__ hfin,  // non-null on last step
    float*      __restrict__ cfin)
{
    extern __shared__ char smem[];                     // 160 KB dynamic
    u16* hLds = (u16*)smem;                            // (64,1024) bf16
    u16* xLds = (u16*)(smem + BATCH * HID * 2);        // (64,256)  bf16
    __shared__ float gbuf[4][BATCH][16];               // 16 KB gate exchange

    const int tid   = threadIdx.x;
    const int wave  = tid >> 5;
    const int lane  = tid & 31;
    const int ln    = lane & 15;
    const int hi    = lane >> 4;           // which K-half of the fragment
    const int mtile = wave & 3;
    const int gate  = wave >> 2;
    const int c0    = blockIdx.x << 4;     // gate-relative column base
    const int nrow  = gate * HID + c0 + ln; // packed-B row (global gate column)

#if HAVE_TDM
    if (wave == 0) {
        tdm_load_1d(hin, hLds, BATCH * HID * 2);
        tdm_load_1d(xb,  xLds, BATCH * ISZ * 2);
        __builtin_amdgcn_s_wait_tensorcnt(0);
    }
#else
    {
        const uint4* hg = (const uint4*)hin;
        uint4*       hl = (uint4*)hLds;
        for (int i = tid; i < BATCH * HID / 8; i += 512) hl[i] = hg[i];
        const uint4* xg = (const uint4*)xb;
        uint4*       xl = (uint4*)xLds;
        for (int i = tid; i < BATCH * ISZ / 8; i += 512) xl[i] = xg[i];
    }
#endif
    __syncthreads();

    v8f acc = {};

    // ---- x contribution: K = 256 ----
    // A (16-bit, 16x32), order used here (same for B): lane<16 holds K {0..15}; lane>=16 holds K {16..31}
    const u16* pa = xLds + (mtile * 16 + ln) * ISZ + hi * 16;
    const u16* pb = WxP + (size_t)nrow * ISZ + hi * 16;  // B: 16 contiguous K per lane
#pragma unroll
    for (int kt = 0; kt < ISZ / 32; ++kt) {
        v16bf a = ld_frag2(pa, pa + 8);
        v16bf b = ld_frag2(pb, pb + 8);
        acc = __builtin_amdgcn_wmma_f32_16x16x32_bf16(false, a, false, b,
                                                      (short)0, acc, false, false);
        pa += 32; pb += 32;
    }

    // ---- h contribution: K = 1024 ----
    pa = hLds + (mtile * 16 + ln) * HID + hi * 16;
    pb = WhP + (size_t)nrow * HID + hi * 16;
#pragma unroll
    for (int kt = 0; kt < HID / 32; ++kt) {
        v16bf a = ld_frag2(pa, pa + 8);
        v16bf b = ld_frag2(pb, pb + 8);
        acc = __builtin_amdgcn_wmma_f32_16x16x32_bf16(false, a, false, b,
                                                      (short)0, acc, false, false);
        pa += 32; pb += 32;
    }

    // C/D layout: VGPR r, lanes 0-15: (M=r, N=lane); lanes 16-31: (M=r+8, N=lane-16)
#pragma unroll
    for (int r = 0; r < 8; ++r)
        gbuf[gate][mtile * 16 + hi * 8 + r][ln] = acc[r];
    __syncthreads();

    // ---- elementwise LSTM cell update: 1024 elems / 512 threads ----
#pragma unroll
    for (int e = tid; e < BATCH * 16; e += 512) {
        int row = e >> 4;
        int cl  = e & 15;
        int col = c0 + cl;
        float gf = fsig(gbuf[0][row][cl] + bias[col]);
        float gi = fsig(gbuf[1][row][cl] + bias[HID + col]);
        float go = fsig(gbuf[2][row][cl] + bias[2 * HID + col]);
        float gc = tanhf(gbuf[3][row][cl] + bias[3 * HID + col]);
        size_t o = (size_t)row * HID + col;
        float cp = Cst[o];
        float cn = gf * cp + gi * gc;
        float hn = go * tanhf(cn);
        Cst[o]  = cn;
        hseq[o] = hn;
        hout[o] = f2bf(hn);
        if (hfin) { hfin[o] = hn; cfin[o] = cn; }
    }
}

// ---------------- host launcher ----------------

extern "C" void kernel_launch(void* const* d_in, const int* in_sizes, int n_in,
                              void* d_out, int out_size, void* d_ws, size_t ws_size,
                              hipStream_t stream) {
    const float* x = (const float*)d_in[0];
    const float* W[9];
    const float* b[9];
    for (int i = 1; i <= 8; ++i) {
        W[i] = (const float*)d_in[2 * i - 1];
        b[i] = (const float*)d_in[2 * i];
    }

    // workspace carve-up (all sizes multiples of 256 B)
    char* ws = (char*)d_ws;
    size_t off = 0;
    u16*   WxP  = (u16*)(ws + off);  off += (size_t)4 * HID * ISZ * sizeof(u16);     // 2 MB
    u16*   WhP  = (u16*)(ws + off);  off += (size_t)4 * HID * HID * sizeof(u16);     // 8 MB
    u16*   xbf  = (u16*)(ws + off);  off += (size_t)SEQ * BATCH * ISZ * sizeof(u16); // 16 MB
    float* bias = (float*)(ws + off); off += (size_t)4 * HID * sizeof(float);        // 16 KB
    float* Cst  = (float*)(ws + off); off += (size_t)BATCH * HID * sizeof(float);    // 256 KB
    u16*   hb0  = (u16*)(ws + off);  off += (size_t)BATCH * HID * sizeof(u16);       // 128 KB
    u16*   hb1  = (u16*)(ws + off);  off += (size_t)BATCH * HID * sizeof(u16);       // 128 KB

    float* out = (float*)d_out;

    // one-time prep (re-run every call: deterministic)
    k_pack<<<(4 * HID * ISZ) / 256, 256, 0, stream>>>(W[1], W[3], W[5], W[7], WxP, ISZ);
    k_pack<<<(4 * HID * HID) / 256, 256, 0, stream>>>(W[2], W[4], W[6], W[8], WhP, HID);
    k_cvt_x<<<(SEQ * BATCH * ISZ) / 256, 256, 0, stream>>>(x, xbf);
    k_prep<<<(BATCH * HID) / 256, 256, 0, stream>>>(b[1], b[2], b[3], b[4],
                                                    b[5], b[6], b[7], b[8],
                                                    bias, Cst, hb0);

    const size_t hc_tail = (size_t)SEQ * BATCH * HID;
    for (int t = 0; t < SEQ; ++t) {
        const u16* hin  = (t & 1) ? hb1 : hb0;
        u16*       hout = (t & 1) ? hb0 : hb1;
        bool last = (t == SEQ - 1);
        lstm_step<<<HID / 16, 512, LDS_DYN_BYTES, stream>>>(
            xbf + (size_t)t * BATCH * ISZ,
            WxP, WhP, bias, hin, hout, Cst,
            out + (size_t)t * BATCH * HID,
            last ? out + hc_tail : nullptr,
            last ? out + hc_tail + (size_t)BATCH * HID : nullptr);
    }
}
